// LoopCoderAttention_6141803233408
// MI455X (gfx1250) — matrix operand order repaired and checked by reading, not compiled, not run on hardware
//
#include <hip/hip_runtime.h>
#include <hip/hip_bf16.h>
#include <cstdint>

#define T_SEQ  2048
#define HID_N  2048
#define HQ_N   16
#define HK_N   4
#define D_HEAD 128
#define WIN_N  64
#define QKV_N  ((HQ_N + 2*HK_N) * D_HEAD)   // 3072
#define SCALE_F 0.08838834764831845f         // 128^-0.5

typedef __attribute__((ext_vector_type(16))) __bf16 v16bf;
typedef __attribute__((ext_vector_type(8)))  __bf16 v8bf;
typedef __attribute__((ext_vector_type(8)))  float  v8f;

static __device__ inline v8f wmma_bf16(v16bf a, v16bf b, v8f c) {
  // D(f32 16x16) = A(bf16 16x32) * B(bf16 32x16) + C
  return __builtin_amdgcn_wmma_f32_16x16x32_bf16(false, a, false, b, (short)0, c,
                                                 false, false);
}

// ---------------------------------------------------------------------------
// Pack f32 MxK row-major -> bf16 A-fragments: out[((mt*Ksteps+ks)*32+lane)*16+i]
// A 16x32 bf16 order used here (same k order as B): lane half hf: element i has K=hf*16+i
// (16 contiguous K), two bf16 per VGPR.
__global__ void pack_a_kernel(const float* __restrict__ A, __bf16* __restrict__ out,
                              int Mtiles, int Ksteps, int ldk) {
  int id = blockIdx.x * blockDim.x + threadIdx.x;
  int total = Mtiles * Ksteps * 32;
  if (id >= total) return;
  int lane = id & 31;
  int c    = id >> 5;
  int ks   = c % Ksteps;
  int mt   = c / Ksteps;
  int mr = lane & 15, hf = lane >> 4;
  const float* arow = A + (size_t)(mt*16 + mr) * ldk + (size_t)ks * 32;
  __bf16* o = out + (size_t)id * 16;
  #pragma unroll
  for (int i = 0; i < 16; ++i) {
    int v = i >> 1, p = i & 1;
    int k = hf*16 + v*2 + p;
    o[i] = (__bf16)arow[k];
  }
}

// Pack f32 KxN row-major -> bf16 B-fragments: out[((nt*Ksteps+ks)*32+lane)*16+i]
// B 32x16 bf16 order used here (same k order as A): lane: N = lane%16, K = hf*16 + i (16 contiguous K).
__global__ void pack_b_kernel(const float* __restrict__ B, __bf16* __restrict__ out,
                              int Ntiles, int Ksteps, int ldn) {
  int id = blockIdx.x * blockDim.x + threadIdx.x;
  int total = Ntiles * Ksteps * 32;
  if (id >= total) return;
  int lane = id & 31;
  int c    = id >> 5;
  int ks   = c % Ksteps;
  int nt   = c / Ksteps;
  int mr = lane & 15, hf = lane >> 4;
  int n = nt*16 + mr;
  __bf16* o = out + (size_t)id * 16;
  #pragma unroll
  for (int i = 0; i < 16; ++i) {
    int k = ks*32 + hf*16 + i;
    o[i] = (__bf16)B[(size_t)k * ldn + n];
  }
}

// ---------------------------------------------------------------------------
// C(f32, MxN) = Apacked @ Bpacked. Register-blocked: each wave computes a
// 16x64 strip (4 N-tiles) reusing one A fragment per K-step. Explicit one-step
// software pipeline: K-step ks+1 fragments load while ks's 4 WMMAs issue.
// Block = 8 waves = 2(M) x 4(N-group) -> 32x256 per block.
__global__ void gemm_kernel(const __bf16* __restrict__ Ap, const __bf16* __restrict__ Bp,
                            float* __restrict__ C, int Ksteps, int ldc) {
  int lane = threadIdx.x & 31;
  int w    = threadIdx.x >> 5;
  int mt  = blockIdx.y * 2 + (w >> 2);
  int nt0 = (blockIdx.x * 4 + (w & 3)) * 4;
  v8f acc[4];
  #pragma unroll
  for (int j = 0; j < 4; ++j)
    #pragma unroll
    for (int r = 0; r < 8; ++r) acc[j][r] = 0.f;
  const size_t bstride = (size_t)Ksteps * 512;
  const __bf16* ap = Ap + (size_t)mt  * bstride + (size_t)lane * 16;
  const __bf16* bp = Bp + (size_t)nt0 * bstride + (size_t)lane * 16;

  v16bf a = *(const v16bf*)ap;
  v16bf b[4];
  #pragma unroll
  for (int j = 0; j < 4; ++j) b[j] = *(const v16bf*)(bp + (size_t)j * bstride);

  for (int ks = 0; ks < Ksteps - 1; ++ks) {
    ap += 512; bp += 512;
    v16bf an = *(const v16bf*)ap;
    v16bf bn[4];
    #pragma unroll
    for (int j = 0; j < 4; ++j) bn[j] = *(const v16bf*)(bp + (size_t)j * bstride);
    #pragma unroll
    for (int j = 0; j < 4; ++j) acc[j] = wmma_bf16(a, b[j], acc[j]);
    a = an;
    #pragma unroll
    for (int j = 0; j < 4; ++j) b[j] = bn[j];
  }
  #pragma unroll
  for (int j = 0; j < 4; ++j) acc[j] = wmma_bf16(a, b[j], acc[j]);

  int mr = lane & 15;
  #pragma unroll
  for (int j = 0; j < 4; ++j) {
    #pragma unroll
    for (int r = 0; r < 8; ++r) {
      int row = mt*16 + ((lane < 16) ? r : r + 8);
      int col = (nt0 + j)*16 + mr;
      C[(size_t)row * ldc + col] = acc[j][r];
    }
  }
}

// ---------------------------------------------------------------------------
// RoPE on q (heads 0..15) and k (heads 16..19 of the fused loop), write bf16.
__global__ void rope_kernel(const float* __restrict__ qkv, const int* __restrict__ pos,
                            __bf16* __restrict__ qb, __bf16* __restrict__ kb) {
  int id = blockIdx.x * blockDim.x + threadIdx.x;
  int j = id & 63;
  int h = (id >> 6) % (HQ_N + HK_N);
  int t = id / (64 * (HQ_N + HK_N));
  if (t >= T_SEQ) return;
  float inv = __powf(10000.0f, -(float)j * (1.0f / 64.0f));
  float ang = (float)pos[t] * inv;
  float s, c;
  __sincosf(ang, &s, &c);
  const float* row = qkv + (size_t)t * QKV_N;
  float x1, x2;
  __bf16 *d1;
  if (h < HQ_N) {
    x1 = row[h*D_HEAD + j];
    x2 = row[h*D_HEAD + 64 + j];
    d1 = qb + ((size_t)t * HQ_N + h) * D_HEAD + j;
  } else {
    int kk = h - HQ_N;
    x1 = row[HQ_N*D_HEAD + kk*D_HEAD + j];
    x2 = row[HQ_N*D_HEAD + kk*D_HEAD + 64 + j];
    d1 = kb + ((size_t)t * HK_N + kk) * D_HEAD + j;
  }
  d1[0]  = (__bf16)(x1 * c - x2 * s);
  d1[64] = (__bf16)(x1 * s + x2 * c);
}

__global__ void conv_bf16_kernel(const float* __restrict__ in, __bf16* __restrict__ out, int n) {
  int id = blockIdx.x * blockDim.x + threadIdx.x;
  if (id < n) out[id] = (__bf16)in[id];
}

// (T,HK,D) slice (row stride ldrow, column offset coloff) -> bf16 (HK,D,T)
__global__ void transpose_v_kernel(const float* __restrict__ src, __bf16* __restrict__ dst,
                                   int ldrow, int coloff) {
  int id = blockIdx.x * blockDim.x + threadIdx.x;
  int d = id % D_HEAD;
  int hk = (id / D_HEAD) % HK_N;
  int t = id / (D_HEAD * HK_N);
  if (t >= T_SEQ) return;
  float v = src[(size_t)t * ldrow + coloff + hk * D_HEAD + d];
  dst[((size_t)hk * D_HEAD + d) * T_SEQ + t] = (__bf16)v;
}

// gate[t,h] = sigmoid( sum_d q[t,h,d]*w_gate[d,h] + b_gate[h] )
__global__ void gate_kernel(const __bf16* __restrict__ qb, const float* __restrict__ wg,
                            const float* __restrict__ bg, float* __restrict__ gate) {
  int id = blockIdx.x * blockDim.x + threadIdx.x;
  int h = id % HQ_N;
  int t = id / HQ_N;
  if (t >= T_SEQ) return;
  const __bf16* q = qb + ((size_t)t * HQ_N + h) * D_HEAD;
  float s = bg[h];
  for (int d = 0; d < D_HEAD; ++d) s += (float)q[d] * wg[d * HQ_N + h];
  gate[id] = 1.0f / (1.0f + __expf(-s));
}

// ---------------------------------------------------------------------------
// One flash step over 32 keys (two 16-key WMMA subtiles).
template <bool LOCAL>
__device__ inline void flash_step(const __bf16* __restrict__ kbase,  // + t*HK*D (head offset applied)
                                  const __bf16* __restrict__ vtbase, // + d*T + t (head offset applied)
                                  const v16bf* qa, int ktbase, int qt, int lane,
                                  float* mrow, float* lrow, v8f* O, __bf16* pls) {
  int mr = lane & 15, hf = lane >> 4;
  v8f s0 = {}, s1 = {};
  const __bf16* k0p = kbase + (size_t)(ktbase + mr)      * (HK_N * D_HEAD);
  const __bf16* k1p = kbase + (size_t)(ktbase + 16 + mr) * (HK_N * D_HEAD);
  #pragma unroll
  for (int ks = 0; ks < 4; ++ks) {
    v16bf b0 = *(const v16bf*)(k0p + ks*32 + hf*16);
    v16bf b1 = *(const v16bf*)(k1p + ks*32 + hf*16);
    s0 = wmma_bf16(qa[ks], b0, s0);
    s1 = wmma_bf16(qa[ks], b1, s1);
  }
  int key0 = ktbase + mr, key1 = ktbase + 16 + mr;
  float p0a[8], p1a[8], alpha[8];
  #pragma unroll
  for (int r = 0; r < 8; ++r) {
    int rowt = qt*16 + ((lane < 16) ? r : r + 8);
    float a = s0[r] * SCALE_F;
    float b = s1[r] * SCALE_F;
    bool v0 = (key0 <= rowt) && (!LOCAL || (rowt - key0) <= WIN_N);
    bool v1 = (key1 <= rowt) && (!LOCAL || (rowt - key1) <= WIN_N);
    a = v0 ? a : -1e30f;
    b = v1 ? b : -1e30f;
    float rm = fmaxf(a, b);
    #pragma unroll
    for (int off = 1; off < 16; off <<= 1) rm = fmaxf(rm, __shfl_xor(rm, off, 32));
    float mnew = fmaxf(mrow[r], rm);
    float al = __expf(mrow[r] - mnew);
    float p0 = __expf(a - mnew);
    float p1 = __expf(b - mnew);
    float rs = p0 + p1;
    #pragma unroll
    for (int off = 1; off < 16; off <<= 1) rs += __shfl_xor(rs, off, 32);
    lrow[r] = lrow[r] * al + rs;
    mrow[r] = mnew;
    alpha[r] = al;
    p0a[r] = p0; p1a[r] = p1;
  }
  #pragma unroll
  for (int j = 0; j < 8; ++j)
    #pragma unroll
    for (int r = 0; r < 8; ++r) O[j][r] *= alpha[r];
  // C-layout -> LDS (row-major 16x32 bf16) -> reload in A-layout
  #pragma unroll
  for (int r = 0; r < 8; ++r) {
    int rowm = (lane < 16) ? r : r + 8;
    pls[rowm*32 + mr]      = (__bf16)p0a[r];
    pls[rowm*32 + 16 + mr] = (__bf16)p1a[r];
  }
  __syncthreads();
  v8bf plo = *(const v8bf*)(pls + mr*32 + hf*16);
  v8bf phi = *(const v8bf*)(pls + mr*32 + hf*16 + 8);
  v16bf pa;
  #pragma unroll
  for (int i = 0; i < 8; ++i) { pa[i] = plo[i]; pa[8+i] = phi[i]; }
  #pragma unroll
  for (int j = 0; j < 8; ++j) {
    v16bf vb = *(const v16bf*)(vtbase + (size_t)(j*16 + mr) * T_SEQ + ktbase + hf*16);
    O[j] = wmma_bf16(pa, vb, O[j]);
  }
  __syncthreads();
}

// grid (HQ, T/16), block 32 (one wave). Fused global+local flash attn + gate blend.
__global__ void attn_kernel(const __bf16* __restrict__ qb,  const __bf16* __restrict__ kgb,
                            const __bf16* __restrict__ vgt, const __bf16* __restrict__ klb,
                            const __bf16* __restrict__ vlt, const float* __restrict__ gate,
                            float* __restrict__ ctx) {
  __shared__ __bf16 pls[16 * 32];
  int h = blockIdx.x, qt = blockIdx.y;
  int lane = threadIdx.x;
  int mr = lane & 15, hf = lane >> 4;
  int kh = h >> 2;  // HQ/HK = 4
  const __bf16* qrow = qb + ((size_t)(qt*16 + mr) * HQ_N + h) * D_HEAD;
  v16bf qa[4];
  #pragma unroll
  for (int ks = 0; ks < 4; ++ks) {
    v8bf lo = *(const v8bf*)(qrow + ks*32 + hf*16);
    v8bf hi = *(const v8bf*)(qrow + ks*32 + hf*16 + 8);
    #pragma unroll
    for (int i = 0; i < 8; ++i) { qa[ks][i] = lo[i]; qa[ks][8+i] = hi[i]; }
  }
  float mg[8], lg[8], ml[8], ll[8];
  v8f Og[8], Ol[8];
  #pragma unroll
  for (int r = 0; r < 8; ++r) { mg[r] = -1e30f; lg[r] = 0.f; ml[r] = -1e30f; ll[r] = 0.f; }
  #pragma unroll
  for (int j = 0; j < 8; ++j)
    #pragma unroll
    for (int r = 0; r < 8; ++r) { Og[j][r] = 0.f; Ol[j][r] = 0.f; }

  int ktmax = (qt*16 + 15) / 32;
  const __bf16* kg0 = kgb + (size_t)kh * D_HEAD;
  const __bf16* vg0 = vgt + (size_t)kh * D_HEAD * T_SEQ;
  for (int kt = 0; kt <= ktmax; ++kt)
    flash_step<false>(kg0, vg0, qa, kt*32, qt, lane, mg, lg, Og, pls);

  int qs = qt * 16;
  int start = (qs >= WIN_N) ? (qs - WIN_N) / 32 : 0;
  const __bf16* kl0 = klb + (size_t)kh * D_HEAD;
  const __bf16* vl0 = vlt + (size_t)kh * D_HEAD * T_SEQ;
  for (int kt = start; kt <= ktmax; ++kt)
    flash_step<true>(kl0, vl0, qa, kt*32, qt, lane, ml, ll, Ol, pls);

  #pragma unroll
  for (int r = 0; r < 8; ++r) {
    int rowt = qt*16 + ((lane < 16) ? r : r + 8);
    float g = gate[(size_t)rowt * HQ_N + h];
    float ig = 1.0f / lg[r], il = 1.0f / ll[r];
    #pragma unroll
    for (int j = 0; j < 8; ++j) {
      float val = (Og[j][r] * ig) * g + (Ol[j][r] * il) * (1.0f - g);
      ctx[(size_t)rowt * (HQ_N * D_HEAD) + h * D_HEAD + j*16 + mr] = val;
    }
  }
}

// ---------------------------------------------------------------------------
extern "C" void kernel_launch(void* const* d_in, const int* in_sizes, int n_in,
                              void* d_out, int out_size, void* d_ws, size_t ws_size,
                              hipStream_t stream) {
  (void)in_sizes; (void)n_in; (void)out_size; (void)ws_size;
  const float* hidden   = (const float*)d_in[0];
  const int*   positions= (const int*)  d_in[1];
  const float* k_global = (const float*)d_in[2];
  const float* v_global = (const float*)d_in[3];
  const float* w_qkv    = (const float*)d_in[4];
  const float* w_o      = (const float*)d_in[5];
  const float* w_gate   = (const float*)d_in[6];
  const float* b_gate   = (const float*)d_in[7];
  float* out = (float*)d_out;

  char* ws = (char*)d_ws;
  size_t off = 0;
  auto carve = [&](size_t bytes) -> char* {
    char* p = ws + off;
    off += (bytes + 255) & ~(size_t)255;
    return p;
  };
  const int Mt = T_SEQ / 16;     // 128
  const int KsH = HID_N / 32;    // 64
  __bf16* hiddenA = (__bf16*)carve((size_t)Mt * KsH * 512 * 2);          // 8 MB
  __bf16* wqkvB   = (__bf16*)carve((size_t)(QKV_N/16) * KsH * 512 * 2);  // 12 MB
  float*  qkv     = (float*) carve((size_t)T_SEQ * QKV_N * 4);           // 24 MB
  __bf16* qb      = (__bf16*)carve((size_t)T_SEQ * HQ_N * D_HEAD * 2);   // 8 MB
  __bf16* kb      = (__bf16*)carve((size_t)T_SEQ * HK_N * D_HEAD * 2);   // 2 MB
  __bf16* kgb     = (__bf16*)carve((size_t)T_SEQ * HK_N * D_HEAD * 2);   // 2 MB
  __bf16* vlt     = (__bf16*)carve((size_t)T_SEQ * HK_N * D_HEAD * 2);   // 2 MB
  __bf16* vgt     = (__bf16*)carve((size_t)T_SEQ * HK_N * D_HEAD * 2);   // 2 MB
  float*  gate    = (float*) carve((size_t)T_SEQ * HQ_N * 4);            // 128 KB
  float*  ctx     = (float*) carve((size_t)T_SEQ * HID_N * 4);           // 16 MB
  __bf16* ctxA    = (__bf16*)carve((size_t)Mt * KsH * 512 * 2);          // 8 MB
  __bf16* woB     = (__bf16*)carve((size_t)(HID_N/16) * KsH * 512 * 2);  // 8 MB

  // 1) QKV projection
  pack_a_kernel<<<(Mt*KsH*32)/256, 256, 0, stream>>>(hidden, hiddenA, Mt, KsH, HID_N);
  pack_b_kernel<<<((QKV_N/16)*KsH*32)/256, 256, 0, stream>>>(w_qkv, wqkvB, QKV_N/16, KsH, QKV_N);
  // block covers 32 rows x 256 cols: grid (N/256, M/32)
  gemm_kernel<<<dim3(QKV_N/256, T_SEQ/32), 256, 0, stream>>>(hiddenA, wqkvB, qkv, KsH, QKV_N);

  // 2) RoPE (q,k) + bf16 conversions / V transposes + gate
  rope_kernel<<<(T_SEQ*(HQ_N+HK_N)*64)/256, 256, 0, stream>>>(qkv, positions, qb, kb);
  conv_bf16_kernel<<<(T_SEQ*HK_N*D_HEAD)/256, 256, 0, stream>>>(k_global, kgb, T_SEQ*HK_N*D_HEAD);
  transpose_v_kernel<<<(T_SEQ*HK_N*D_HEAD)/256, 256, 0, stream>>>(v_global, vgt, HK_N*D_HEAD, 0);
  transpose_v_kernel<<<(T_SEQ*HK_N*D_HEAD)/256, 256, 0, stream>>>(qkv, vlt, QKV_N, (HQ_N+HK_N)*D_HEAD);
  gate_kernel<<<(T_SEQ*HQ_N)/256, 256, 0, stream>>>(qb, w_gate, b_gate, gate);

  // 3) Fused dual flash attention + gating
  attn_kernel<<<dim3(HQ_N, T_SEQ/16), 32, 0, stream>>>(qb, kgb, vgt, kb, vlt, gate, ctx);

  // 4) Output projection
  pack_a_kernel<<<(Mt*KsH*32)/256, 256, 0, stream>>>(ctx, ctxA, Mt, KsH, HID_N);
  pack_b_kernel<<<((HID_N/16)*KsH*32)/256, 256, 0, stream>>>(w_o, woB, HID_N/16, KsH, HID_N);
  gemm_kernel<<<dim3(HID_N/256, T_SEQ/32), 256, 0, stream>>>(ctxA, woB, out, KsH, HID_N);
}
